// RNN_80599356276959
// MI455X (gfx1250) — matrix operand order repaired and checked by reading, not compiled, not run on hardware
//
#include <hip/hip_runtime.h>

// ---------------------------------------------------------------------------
// RNN forward on gfx1250 (MI455X), bf16 WMMA pipeline:
//   K1 prepare:   W_xh^T, W_hh^T -> bf16 (contiguous-K for B fragments); h0=0
//   K1b gather:   xe = bf16(emb[x])  (convert once, not per N-supertile)
//   K2 GEMM:      xw[b*s][h] = bf16( xe @ W_xh + b_h )        (WMMA bf16)
//   K3 recurrence: persistent kernel, 512 steps, W_hh slice in LDS (async),
//                  grid-wide atomic barrier per step, h = tanh(xw_t + h@W_hh),
//                  prefetch of next step's xw slice
//   K4 classifier: out = h_last @ cls_w + cls_bias  (cls_w f32 -> bf16 in-reg)
// ---------------------------------------------------------------------------

typedef __attribute__((ext_vector_type(16))) __bf16         v16bf;
typedef __attribute__((ext_vector_type(2)))  __bf16         v2bf;
typedef __attribute__((ext_vector_type(16))) unsigned short v16us;
typedef __attribute__((ext_vector_type(8)))  float          v8f;
typedef __attribute__((ext_vector_type(4)))  unsigned int   v4u;
typedef __attribute__((ext_vector_type(4)))  int            v4i;

#define BB  64
#define SS  512
#define DD  512
#define HH  1024
#define OO  32000

#define GLOBAL_AS __attribute__((address_space(1)))
#define LDS_AS    __attribute__((address_space(3)))

#if defined(__has_builtin)
#  if __has_builtin(__builtin_amdgcn_cvt_pk_bf16_f32)
#    define HAVE_CVT_PK 1
#  else
#    define HAVE_CVT_PK 0
#  endif
#  if __has_builtin(__builtin_amdgcn_global_load_async_to_lds_b128)
#    define HAVE_ASYNC_LDS 1
#  else
#    define HAVE_ASYNC_LDS 0
#  endif
#  if __has_builtin(__builtin_amdgcn_s_wait_asynccnt)
#    define HAVE_WAIT_ASYNC 1
#  else
#    define HAVE_WAIT_ASYNC 0
#  endif
#else
#  define HAVE_CVT_PK 0
#  define HAVE_ASYNC_LDS 0
#  define HAVE_WAIT_ASYNC 0
#endif

union Frag {
  v16bf    bf;
  v16us    us;
  unsigned d8[8];
  v4u      q[2];
};

// f32 -> bf16, round-half-up (ties differ from RNE only): 1 add + shift
__device__ __forceinline__ unsigned short f2bf(float f) {
  return (unsigned short)((__float_as_uint(f) + 0x8000u) >> 16);
}
__device__ __forceinline__ float bf2f(unsigned short h) {
  return __uint_as_float(((unsigned)h) << 16);
}
// pack two f32 -> two bf16 in one dword. v_cvt_pk_bf16_f32 when available,
// else 2x v_add + 1x v_perm_b32 (splice high halves: a -> [15:0], b -> [31:16])
__device__ __forceinline__ unsigned cvt2(float a, float b) {
#if HAVE_CVT_PK
  v2bf p = __builtin_amdgcn_cvt_pk_bf16_f32(a, b);
  return __builtin_bit_cast(unsigned, p);
#else
  unsigned ua = __float_as_uint(a) + 0x8000u;
  unsigned ub = __float_as_uint(b) + 0x8000u;
  // {S0,S1}=bytes[7..4,3..0]; select ua.b2,ua.b3,ub.b2,ub.b3 (LSB first)
  return __builtin_amdgcn_perm(ub, ua, 0x07060302u);
#endif
}

__device__ __forceinline__ v8f wmma_bf16(Frag a, Frag b, v8f c) {
  // D = A(16x32 bf16) x B(32x16 bf16) + C(16x16 f32)
  return __builtin_amdgcn_wmma_f32_16x16x32_bf16(
      /*neg_a=*/false, a.bf, /*neg_b=*/false, b.bf,
      /*c_mod=*/(short)0, c, /*reuse_a=*/false, /*reuse_b=*/false);
}

__device__ __forceinline__ v8f vzero8() {
  v8f z = {0.f, 0.f, 0.f, 0.f, 0.f, 0.f, 0.f, 0.f};
  return z;
}

// ---------------------------------------------------------------------------
// K1: transpose+convert weights to bf16, zero h0, reset grid barrier counter.
// W^T layout gives each WMMA B-lane 16 *contiguous* K values -> 2x b128 loads.
// ---------------------------------------------------------------------------
__global__ __launch_bounds__(256)
void k_prepare(const float* __restrict__ Wxh, const float* __restrict__ Whh,
               unsigned short* __restrict__ wxhT, unsigned short* __restrict__ whhT,
               unsigned short* __restrict__ h0, unsigned* __restrict__ cnt) {
  unsigned idx = blockIdx.x * blockDim.x + threadIdx.x;
  if (idx < (unsigned)(HH * HH)) {                 // W_hh: [K=H][N=H] -> [N][K]
    unsigned n = idx / HH, k = idx % HH;           // consecutive idx -> coalesced write
    whhT[(size_t)n * HH + k] = f2bf(Whh[(size_t)k * HH + n]);
  }
  if (idx < (unsigned)(HH * DD)) {                 // W_xh: [K=D][N=H] -> [N][K]
    unsigned n = idx / DD, k = idx % DD;
    wxhT[(size_t)n * DD + k] = f2bf(Wxh[(size_t)k * HH + n]);
  }
  if (idx < (unsigned)(BB * HH)) h0[idx] = 0;      // bf16 zero
  if (idx == 0) *cnt = 0;                          // grid-barrier counter
}

// ---------------------------------------------------------------------------
// K1b: xe[row][d] = bf16(emb[x[row]][d]). One pair-convert per thread,
// fully coalesced; removes all conversion work from K2's inner loop.
// ---------------------------------------------------------------------------
__global__ __launch_bounds__(256)
void k_gather(const int* __restrict__ x, const float* __restrict__ emb,
              unsigned* __restrict__ xe_pairs) {   // xe as dword pairs
  const unsigned idx = blockIdx.x * blockDim.x + threadIdx.x;  // pair index
  const unsigned row = idx >> 8;                   // DD/2 = 256 pairs per row
  const unsigned kp  = (idx & 255u) << 1;
  const int tok = x[row];
  const float* src = emb + (size_t)tok * DD + kp;
  xe_pairs[idx] = cvt2(src[0], src[1]);
}

// ---------------------------------------------------------------------------
// K2: xw = xe @ W_xh + b_h, M=32768 N=1024 K=512 (all-bf16 operands).
// One wave per 16x64 output strip; A and B are pure b128 loads.
// ---------------------------------------------------------------------------
__global__ __launch_bounds__(256)
void k_embed_gemm(const unsigned short* __restrict__ xe,
                  const unsigned short* __restrict__ wxhT,
                  const float* __restrict__ bh, unsigned short* __restrict__ xw) {
  const int lane = threadIdx.x & 31;
  const int hf   = lane >> 4;        // lane half: selects K sub-ranges
  const int lm   = lane & 15;
  const unsigned w     = blockIdx.x * (blockDim.x >> 5) + (threadIdx.x >> 5);
  const unsigned tileM = w & 2047;   // 2048 M-tiles of 16 rows
  const unsigned n0    = (w >> 11) * 64;  // 16 N-supertiles of 64 cols
  const unsigned short* arow = xe + (size_t)(tileM * 16 + lm) * DD;

  v8f acc[4] = {vzero8(), vzero8(), vzero8(), vzero8()};

  for (int kb = 0; kb < DD; kb += 32) {
    Frag a;
    a.q[0] = *(const v4u*)(arow + kb + 16 * hf);
    a.q[1] = *(const v4u*)(arow + kb + 16 * hf + 8);
#pragma unroll
    for (int nt = 0; nt < 4; ++nt) {
      const unsigned short* brow =
          wxhT + (size_t)(n0 + nt * 16 + lm) * DD + kb + 16 * hf;
      Frag b;
      b.q[0] = *(const v4u*)(brow);
      b.q[1] = *(const v4u*)(brow + 8);
      acc[nt] = wmma_bf16(a, b, acc[nt]);
    }
  }
#pragma unroll
  for (int nt = 0; nt < 4; ++nt) {
    const unsigned n = n0 + nt * 16 + lm;
    const float bias = bh[n];
#pragma unroll
    for (int r = 0; r < 8; ++r) {
      const unsigned row = tileM * 16 + 8 * hf + r;   // C layout: m = r + 8*half
      xw[(size_t)row * HH + n] = f2bf(acc[nt][r] + bias);
    }
  }
}

// ---------------------------------------------------------------------------
// K3: persistent recurrence. 32 blocks x 8 waves; block owns 32 columns of H.
// Its W_hh^T slice (32x1024 bf16 = 64KB) stays in LDS for ALL 512 steps
// (staged with async-to-LDS). Per step: each wave does a 16x16 tile, K=1024
// (32 bf16 WMMAs), prefetches next step's xw, tanh epilogue, then a
// device-wide monotonic atomic barrier (h is globally shared).
// ---------------------------------------------------------------------------
__global__ __launch_bounds__(256)
void k_rnn(const unsigned short* __restrict__ whhT,
           const unsigned short* __restrict__ xw,
           unsigned short* __restrict__ h0, unsigned short* __restrict__ h1,
           unsigned* __restrict__ cnt) {
  __shared__ unsigned short ldsB[32 * HH];   // 64 KB persistent B slice

  const int tid   = threadIdx.x;
  const int lane  = tid & 31;
  const int hf    = lane >> 4;
  const int lm    = lane & 15;
  const int wv    = tid >> 5;          // 0..7
  const int tileM = wv & 3;            // 4 tiles cover M=64
  const int tN    = wv >> 2;           // 0..1 -> 2 tiles cover 32 cols
  const unsigned n0 = blockIdx.x * 32;

  { // stage W_hh^T rows [n0, n0+32) into LDS once
    const unsigned short* src = whhT + (size_t)n0 * HH;
#if HAVE_ASYNC_LDS
    for (int i = tid; i < (32 * HH) / 8; i += (int)blockDim.x) {
      GLOBAL_AS v4i* gsrc =
          (GLOBAL_AS v4i*)(unsigned long long)(src + (size_t)i * 8);
      LDS_AS v4i* ldst =
          (LDS_AS v4i*)(unsigned)(unsigned long long)(ldsB + (size_t)i * 8);
      __builtin_amdgcn_global_load_async_to_lds_b128(gsrc, ldst, 0, 0);
    }
#if HAVE_WAIT_ASYNC
    __builtin_amdgcn_s_wait_asynccnt(0);
#else
    asm volatile("s_wait_asynccnt 0" ::: "memory");
#endif
#else
    const v4u* s4 = (const v4u*)src;
    v4u* dst = (v4u*)ldsB;
    for (int i = tid; i < (32 * HH) / 8; i += (int)blockDim.x) dst[i] = s4[i];
#endif
  }
  __syncthreads();

  const unsigned short* brow = ldsB + (size_t)(tN * 16 + lm) * HH;
  const int m      = tileM * 16 + lm;
  const unsigned n = n0 + tN * 16 + lm;

  for (int t = 0; t < SS; ++t) {
    const unsigned short* hc = (t & 1) ? h1 : h0;   // t even: read h0, write h1
    unsigned short*       hn = (t & 1) ? h0 : h1;   // t=511 writes h0 => h_last
    const unsigned short* arow = hc + (size_t)m * HH;

    if (t + 1 < SS) {                  // hide HBM latency of next xw slice
#pragma unroll
      for (int r = 0; r < 8; ++r) {
        const int bidx = tileM * 16 + 8 * hf + r;
        __builtin_prefetch(&xw[((size_t)bidx * SS + (t + 1)) * HH + n], 0, 1);
      }
    }

    v8f c = vzero8();
#pragma unroll 8
    for (int kb = 0; kb < HH; kb += 32) {
      Frag a, b;
      a.q[0] = *(const v4u*)(arow + kb + 16 * hf);
      a.q[1] = *(const v4u*)(arow + kb + 16 * hf + 8);
      b.q[0] = *(const v4u*)(brow + kb + 16 * hf);        // ds_load_b128
      b.q[1] = *(const v4u*)(brow + kb + 16 * hf + 8);
      c = wmma_bf16(a, b, c);
    }
#pragma unroll
    for (int r = 0; r < 8; ++r) {
      const int bidx = tileM * 16 + 8 * hf + r;           // batch row
      const float xv = bf2f(xw[((size_t)bidx * SS + t) * HH + n]);
      hn[(size_t)bidx * HH + n] = f2bf(tanhf(c[r] + xv));
    }

    // ---- device-wide barrier (monotonic counter; reset by k_prepare) ----
    __threadfence();
    __syncthreads();
    if (tid == 0) {
      __hip_atomic_fetch_add(cnt, 1u, __ATOMIC_ACQ_REL, __HIP_MEMORY_SCOPE_AGENT);
      const unsigned goal = (unsigned)(t + 1) * gridDim.x;
      while (__hip_atomic_load(cnt, __ATOMIC_ACQUIRE, __HIP_MEMORY_SCOPE_AGENT) < goal)
        __builtin_amdgcn_s_sleep(1);
    }
    __syncthreads();
    __threadfence();
  }
}

// ---------------------------------------------------------------------------
// K4: out = h_last @ cls_w + cls_bias. HBM-bound on cls_w (128MB f32, read
// once, converted to bf16 in-register; coalesced across the 16 column lanes).
// One 16x16 tile per wave -> 8000 waves for memory-latency hiding.
// ---------------------------------------------------------------------------
__global__ __launch_bounds__(256)
void k_classifier(const unsigned short* __restrict__ hlast,
                  const float* __restrict__ clsw, const float* __restrict__ clsb,
                  float* __restrict__ out) {
  const int lane = threadIdx.x & 31;
  const int hf   = lane >> 4;
  const int lm   = lane & 15;
  const unsigned w     = blockIdx.x * (blockDim.x >> 5) + (threadIdx.x >> 5);
  const unsigned tileM = w & 3;          // 4 M-tiles
  const unsigned n     = (w >> 2) * 16 + lm;   // 2000 N-tiles of 16

  const unsigned short* arow = hlast + (size_t)(tileM * 16 + lm) * HH;
  const float* bcol = clsw + n;          // column base, stride OO between K
  v8f acc = vzero8();

  for (int kb = 0; kb < HH; kb += 32) {
    Frag a, b;
    a.q[0] = *(const v4u*)(arow + kb + 16 * hf);
    a.q[1] = *(const v4u*)(arow + kb + 16 * hf + 8);
    const int k0 = kb + 16 * hf;         // 16 consecutive K for this lane
#pragma unroll
    for (int p = 0; p < 8; ++p)
      b.d8[p] = cvt2(bcol[(size_t)(k0 + 2 * p) * OO],
                     bcol[(size_t)(k0 + 2 * p + 1) * OO]);
    acc = wmma_bf16(a, b, acc);
  }
  const float bias = clsb[n];
#pragma unroll
  for (int r = 0; r < 8; ++r) {
    const unsigned row = tileM * 16 + 8 * hf + r;
    out[(size_t)row * OO + n] = acc[r] + bias;
  }
}

// ---------------------------------------------------------------------------
extern "C" void kernel_launch(void* const* d_in, const int* in_sizes, int n_in,
                              void* d_out, int out_size, void* d_ws, size_t ws_size,
                              hipStream_t stream) {
  const int*   x    = (const int*)d_in[0];
  const float* emb  = (const float*)d_in[1];
  const float* Wxh  = (const float*)d_in[2];
  const float* Whh  = (const float*)d_in[3];
  const float* bh   = (const float*)d_in[4];
  const float* clsw = (const float*)d_in[5];
  const float* clsb = (const float*)d_in[6];
  float* out = (float*)d_out;

  // workspace layout (bytes)
  char* ws = (char*)d_ws;
  unsigned short* xw   = (unsigned short*)(ws);                    //  64 MB [B*S][H]  bf16
  unsigned short* xe   = (unsigned short*)(ws + 67108864ull);      //  32 MB [B*S][D]  bf16
  unsigned short* whhT = (unsigned short*)(ws + 100663296ull);     //   2 MB [H][H]    bf16
  unsigned short* wxhT = (unsigned short*)(ws + 102760448ull);     //   1 MB [H][D]    bf16
  unsigned short* h0   = (unsigned short*)(ws + 103809024ull);     // 128 KB [B][H]    bf16
  unsigned short* h1   = (unsigned short*)(ws + 103940096ull);     // 128 KB
  unsigned*       cnt  = (unsigned*)(ws + 104071168ull);           //   4 B barrier

  hipLaunchKernelGGL(k_prepare,    dim3(4096),  dim3(256), 0, stream,
                     Wxh, Whh, wxhT, whhT, h0, cnt);
  hipLaunchKernelGGL(k_gather,     dim3(32768), dim3(256), 0, stream,
                     x, emb, (unsigned*)xe);
  hipLaunchKernelGGL(k_embed_gemm, dim3(4096),  dim3(256), 0, stream,
                     xe, wxhT, bh, xw);
  hipLaunchKernelGGL(k_rnn,        dim3(32),    dim3(256), 0, stream,
                     whhT, xw, h0, h1, cnt);
  hipLaunchKernelGGL(k_classifier, dim3(1000),  dim3(256), 0, stream,
                     h0, clsw, clsb, out);   // t=511 wrote h0 -> h_last
}
